// DecoderRNNCellJointCopy_38792144618062
// MI455X (gfx1250) — compile-verified, matrix operand order checked by reading, not run on hardware
//
#include <hip/hip_runtime.h>
#include <hip/hip_bf16.h>
#include <math.h>

// ---------------------------------------------------------------------------
// Sizes
#define Bz 16
#define Tz 600
#define Vz 20000
#define Ez 300
#define Hz 512
#define XDIM1 (Ez + Hz)      // 812
#define G4H (4 * Hz)         // 2048
#define BT (Bz * Tz)         // 9600
#define N2H (2 * Hz)         // 1024 (two attention heads fused)

typedef __attribute__((ext_vector_type(16))) __bf16 v16bf;
typedef __attribute__((ext_vector_type(8)))  __bf16 v8bf;
typedef __attribute__((ext_vector_type(8)))  float  v8f;

// ---------------------------------------------------------------------------
// K0: emb_att = concat(emb_table[x], last_hidden_attn)   (16 x 812)
__global__ void emb_kernel(const int* __restrict__ x,
                           const float* __restrict__ emb_table,
                           const float* __restrict__ lha,
                           float* __restrict__ emb_att) {
    int i = blockIdx.x * blockDim.x + threadIdx.x;
    if (i >= Bz * XDIM1) return;
    int c = i % XDIM1, b = i / XDIM1;
    emb_att[i] = (c < Ez) ? emb_table[(size_t)x[b] * Ez + c]
                          : lha[b * Hz + (c - Ez)];
}

// ---------------------------------------------------------------------------
// K1: f32 -> bf16 elementwise (enc_outs copy for WMMA)
__global__ void cvt_bf16_kernel(const float* __restrict__ in,
                                __bf16* __restrict__ out, int n) {
    int i = blockIdx.x * blockDim.x + threadIdx.x;
    if (i < n) out[i] = (__bf16)in[i];
}

// K2: Wt[n][k] = (n<512 ? aw2g[k][n] : aw2c[k][n-512])  -> bf16, N-major
__global__ void build_wt_kernel(const float* __restrict__ aw2g,
                                const float* __restrict__ aw2c,
                                __bf16* __restrict__ Wt) {
    int i = blockIdx.x * blockDim.x + threadIdx.x;
    if (i >= N2H * Hz) return;
    int n = i >> 9, k = i & (Hz - 1);
    float v = (n < Hz) ? aw2g[k * Hz + n] : aw2c[k * Hz + (n - Hz)];
    Wt[(size_t)n * Hz + k] = (__bf16)v;
}

// ---------------------------------------------------------------------------
// K3: LSTM cell (generic): thread per (b, j)
__global__ void lstm_kernel(const float* __restrict__ xin, int xdim,
                            const float* __restrict__ hprev,
                            const float* __restrict__ cprev,
                            const float* __restrict__ K,
                            const float* __restrict__ R,
                            const float* __restrict__ bias,
                            float* __restrict__ hout,
                            float* __restrict__ cout) {
    int i = blockIdx.x * blockDim.x + threadIdx.x;
    if (i >= Bz * Hz) return;
    int j = i & (Hz - 1), b = i >> 9;
    const float* xr = xin + (size_t)b * xdim;
    const float* hr = hprev + (size_t)b * Hz;
    float z[4];
#pragma unroll
    for (int g = 0; g < 4; ++g) {
        int col = g * Hz + j;
        float acc = bias[col];
        for (int k = 0; k < xdim; ++k) acc += xr[k] * K[(size_t)k * G4H + col];
        for (int k = 0; k < Hz;   ++k) acc += hr[k] * R[(size_t)k * G4H + col];
        z[g] = acc;
    }
    float ig = 1.f / (1.f + __expf(-z[0]));
    float fg = 1.f / (1.f + __expf(-z[1]));
    float gg = tanhf(z[2]);
    float og = 1.f / (1.f + __expf(-z[3]));
    float cn = fg * cprev[i] + ig * gg;
    cout[i] = cn;
    hout[i] = og * tanhf(cn);
}

// ---------------------------------------------------------------------------
// K4: q[head][b][col] = h2n[b] @ aw1{g,c}
__global__ void qproj_kernel(const float* __restrict__ h2n,
                             const float* __restrict__ aw1g,
                             const float* __restrict__ aw1c,
                             float* __restrict__ q) {
    int i = blockIdx.x * blockDim.x + threadIdx.x;
    if (i >= 2 * Bz * Hz) return;
    int col = i & (Hz - 1);
    int b = (i >> 9) & (Bz - 1);
    int head = i >> 13;
    const float* W = head ? aw1c : aw1g;
    const float* hr = h2n + b * Hz;
    float acc = 0.f;
    for (int k = 0; k < Hz; ++k) acc += hr[k] * W[(size_t)k * Hz + col];
    q[i] = acc;
}

// ---------------------------------------------------------------------------
// K5: WMMA GEMM.  S(9600 x 1024) = enc_bf(9600 x 512) @ Wt^T
// Each wave: 32 rows (2 M tiles) x 64 cols (4 N tiles) = 8 accumulators.
// All 6 fragment-load pairs per K-step go to DISTINCT registers so the
// compiler can clause them and issue partial waits instead of full drains.
__global__ __launch_bounds__(128)
void attn_gemm_wmma(const __bf16* __restrict__ A,   // BT x Hz, row-major
                    const __bf16* __restrict__ Bt,  // N2H x Hz, N-major
                    float* __restrict__ S) {        // BT x N2H
    const int lane  = threadIdx.x & 31;
    const int wave  = (blockIdx.x << 2) + (threadIdx.x >> 5);   // 4800 waves
    const int m0    = (wave >> 4) << 5;        // 300 M groups of 32 rows
    const int n0    = (wave & 15) << 6;        // 16 N groups of 64 cols
    const int lrow  = lane & 15;
    const int kSel  = (lane >> 4) << 3;        // 0 or 8 (ISA bf16 frag layout)

    v8f acc[2][4] = {};
    const __bf16* aBase = A  + (size_t)(m0 + lrow) * Hz + kSel;
    const __bf16* bBase = Bt + (size_t)(n0 + lrow) * Hz + kSel;

    for (int k0 = 0; k0 < Hz; k0 += 32) {
        v16bf afr[2];
#pragma unroll
        for (int mi = 0; mi < 2; ++mi) {
            const __bf16* p = aBase + (size_t)mi * 16 * Hz + k0;
            v8bf lo = *(const v8bf*)p;
            v8bf hi = *(const v8bf*)(p + 16);
#pragma unroll
            for (int e = 0; e < 8; ++e) { afr[mi][e] = lo[e]; afr[mi][e + 8] = hi[e]; }
        }
        v16bf bfr[4];
#pragma unroll
        for (int nj = 0; nj < 4; ++nj) {
            const __bf16* p = bBase + (size_t)nj * 16 * Hz + k0;
            v8bf lo = *(const v8bf*)p;
            v8bf hi = *(const v8bf*)(p + 16);
#pragma unroll
            for (int e = 0; e < 8; ++e) { bfr[nj][e] = lo[e]; bfr[nj][e + 8] = hi[e]; }
        }
#pragma unroll
        for (int mi = 0; mi < 2; ++mi)
#pragma unroll
            for (int nj = 0; nj < 4; ++nj)
                acc[mi][nj] = __builtin_amdgcn_wmma_f32_16x16x32_bf16(
                    false, afr[mi], false, bfr[nj], (short)0, acc[mi][nj],
                    false, false);
    }
    // D layout: vgpr r, lane l -> (m = tile_m + r + 8*(l>>4), n = tile_n + (l&15))
    const int mHi = (lane >> 4) << 3;
#pragma unroll
    for (int mi = 0; mi < 2; ++mi)
#pragma unroll
        for (int nj = 0; nj < 4; ++nj)
#pragma unroll
            for (int r = 0; r < 8; ++r)
                S[(size_t)(m0 + mi * 16 + mHi + r) * N2H + n0 + nj * 16 + lrow] =
                    acc[mi][nj][r];
}

// ---------------------------------------------------------------------------
// K6: per (b, head): score -> softmax -> context (head 0) / align (head 1)
__global__ __launch_bounds__(256)
void attention_kernel(const float* __restrict__ S,
                      const float* __restrict__ q,
                      const float* __restrict__ avg,
                      const float* __restrict__ avc,
                      const float* __restrict__ enc_outs,
                      float* __restrict__ context,
                      float* __restrict__ align_c) {
    __shared__ float sc[Tz];
    __shared__ float red[256];
    __shared__ float sq[Hz];
    __shared__ float sv[Hz];
    const int b = blockIdx.x & (Bz - 1);
    const int head = blockIdx.x >> 4;
    const int tid = threadIdx.x;
    const float* qrow = q + (size_t)(head * Bz + b) * Hz;
    const float* vvec = head ? avc : avg;
    for (int h = tid; h < Hz; h += 256) { sq[h] = qrow[h]; sv[h] = vvec[h]; }
    __syncthreads();

    for (int t = tid; t < Tz; t += 256) {
        const float* srow = S + (size_t)(b * Tz + t) * N2H + head * Hz;
        float acc = 0.f;
        for (int h = 0; h < Hz; ++h) acc += tanhf(sq[h] + srow[h]) * sv[h];
        sc[t] = acc;
    }
    __syncthreads();
    // softmax over T
    float m = -INFINITY;
    for (int t = tid; t < Tz; t += 256) m = fmaxf(m, sc[t]);
    red[tid] = m; __syncthreads();
    for (int s = 128; s > 0; s >>= 1) {
        if (tid < s) red[tid] = fmaxf(red[tid], red[tid + s]);
        __syncthreads();
    }
    m = red[0]; __syncthreads();
    float sum = 0.f;
    for (int t = tid; t < Tz; t += 256) { float e = __expf(sc[t] - m); sc[t] = e; sum += e; }
    red[tid] = sum; __syncthreads();
    for (int s = 128; s > 0; s >>= 1) {
        if (tid < s) red[tid] += red[tid + s];
        __syncthreads();
    }
    const float inv = 1.f / red[0];
    __syncthreads();

    if (head) {
        for (int t = tid; t < Tz; t += 256) align_c[b * Tz + t] = sc[t] * inv;
    } else {
        for (int d = tid; d < Hz; d += 256) {
            float acc = 0.f;
            for (int t = 0; t < Tz; ++t)
                acc += sc[t] * enc_outs[((size_t)b * Tz + t) * Hz + d];
            context[b * Hz + d] = acc * inv;
        }
    }
}

// ---------------------------------------------------------------------------
// K7: hidden_att = tanh(concat(context, h2n) @ Wh + bh)
__global__ void hidden_att_kernel(const float* __restrict__ context,
                                  const float* __restrict__ h2n,
                                  const float* __restrict__ Wh,
                                  const float* __restrict__ bh,
                                  float* __restrict__ ha) {
    int i = blockIdx.x * blockDim.x + threadIdx.x;
    if (i >= Bz * Hz) return;
    int j = i & (Hz - 1), b = i >> 9;
    float acc = bh[j];
    const float* cr = context + b * Hz;
    const float* hr = h2n + b * Hz;
    for (int k = 0; k < Hz; ++k) acc += cr[k] * Wh[(size_t)k * Hz + j];
    for (int k = 0; k < Hz; ++k) acc += hr[k] * Wh[(size_t)(Hz + k) * Hz + j];
    ha[i] = tanhf(acc);
}

// K8: switch = sigmoid(ha @ Wf + bf)
__global__ void switch_kernel(const float* __restrict__ ha,
                              const float* __restrict__ Wf,
                              const float* __restrict__ bfp,
                              float* __restrict__ sw) {
    __shared__ float red[256];
    int b = blockIdx.x, tid = threadIdx.x;
    float acc = 0.f;
    for (int k = tid; k < Hz; k += 256) acc += ha[b * Hz + k] * Wf[k];
    red[tid] = acc; __syncthreads();
    for (int s = 128; s > 0; s >>= 1) {
        if (tid < s) red[tid] += red[tid + s];
        __syncthreads();
    }
    if (tid == 0) sw[b] = 1.f / (1.f + __expf(-(red[0] + bfp[0])));
}

// ---------------------------------------------------------------------------
// K9: gen logits (16 batch accumulators per thread; Wg streamed once)
__global__ __launch_bounds__(128)
void gen_logits_kernel(const float* __restrict__ ha,
                       const float* __restrict__ Wg,
                       const float* __restrict__ bg,
                       float* __restrict__ gl) {
    __shared__ float sha[Bz * Hz];
    int tid = threadIdx.x;
    for (int i = tid; i < Bz * Hz; i += 128) sha[i] = ha[i];
    __syncthreads();
    int v = blockIdx.x * 128 + tid;
    if (v >= Vz) return;
    float acc[Bz];
#pragma unroll
    for (int b = 0; b < Bz; ++b) acc[b] = 0.f;
    for (int k = 0; k < Hz; ++k) {
        float w = Wg[(size_t)k * Vz + v];
#pragma unroll
        for (int b = 0; b < Bz; ++b) acc[b] += sha[b * Hz + k] * w;
    }
    float bgv = bg[v];
#pragma unroll
    for (int b = 0; b < Bz; ++b) gl[(size_t)b * Vz + v] = acc[b] + bgv;
}

// K10: per-row max / sum-exp over V
__global__ void softmax_stats_kernel(const float* __restrict__ gl,
                                     float* __restrict__ rowmax,
                                     float* __restrict__ rowsum) {
    __shared__ float red[256];
    int b = blockIdx.x, tid = threadIdx.x;
    const float* row = gl + (size_t)b * Vz;
    float m = -INFINITY;
    for (int v = tid; v < Vz; v += 256) m = fmaxf(m, row[v]);
    red[tid] = m; __syncthreads();
    for (int s = 128; s > 0; s >>= 1) {
        if (tid < s) red[tid] = fmaxf(red[tid], red[tid + s]);
        __syncthreads();
    }
    m = red[0]; __syncthreads();
    float sum = 0.f;
    for (int v = tid; v < Vz; v += 256) sum += __expf(row[v] - m);
    red[tid] = sum; __syncthreads();
    for (int s = 128; s > 0; s >>= 1) {
        if (tid < s) red[tid] += red[tid + s];
        __syncthreads();
    }
    if (tid == 0) { rowmax[b] = m; rowsum[b] = red[0]; }
}

// ---------------------------------------------------------------------------
// K11: copy_prob einsum (768 MB stream) fused with final mix.
__global__ __launch_bounds__(256)
void final_kernel(const float* __restrict__ gl,
                  const float* __restrict__ rowmax,
                  const float* __restrict__ rowsum,
                  const float* __restrict__ sw,
                  const float* __restrict__ align_c,
                  const float* __restrict__ enc_ins,
                  float* __restrict__ result) {
    __shared__ float sal[Tz];
    const int b = blockIdx.y;
    const int tid = threadIdx.x;
    for (int t = tid; t < Tz; t += 256) sal[t] = align_c[b * Tz + t];
    __syncthreads();
    int v = blockIdx.x * 256 + tid;
    if (v >= Vz) return;
    float acc = 0.f;
    const float* base = enc_ins + (size_t)b * Tz * Vz + v;
    for (int t = 0; t < Tz; ++t) acc += sal[t] * base[(size_t)t * Vz];
    float g = __expf(gl[(size_t)b * Vz + v] - rowmax[b]) / rowsum[b];
    float s = sw[b];
    result[(size_t)b * Vz + v] = g * (1.f - s) + acc * s;
}

// ---------------------------------------------------------------------------
extern "C" void kernel_launch(void* const* d_in, const int* in_sizes, int n_in,
                              void* d_out, int out_size, void* d_ws, size_t ws_size,
                              hipStream_t stream) {
    const int*   x        = (const int*)  d_in[0];
    const float* lha      = (const float*)d_in[1];
    const float* h1       = (const float*)d_in[2];
    const float* c1       = (const float*)d_in[3];
    const float* h2       = (const float*)d_in[4];
    const float* c2       = (const float*)d_in[5];
    const float* enc_outs = (const float*)d_in[6];
    const float* enc_ins  = (const float*)d_in[7];
    const float* emb_tab  = (const float*)d_in[8];
    const float* k1       = (const float*)d_in[9];
    const float* r1       = (const float*)d_in[10];
    const float* b1       = (const float*)d_in[11];
    const float* k2       = (const float*)d_in[12];
    const float* r2       = (const float*)d_in[13];
    const float* b2       = (const float*)d_in[14];
    const float* Wh       = (const float*)d_in[15];
    const float* bh       = (const float*)d_in[16];
    const float* Wg       = (const float*)d_in[17];
    const float* bg       = (const float*)d_in[18];
    const float* Wf       = (const float*)d_in[19];
    const float* bfp      = (const float*)d_in[20];
    const float* aw1g     = (const float*)d_in[21];
    const float* aw2g     = (const float*)d_in[22];
    const float* avg      = (const float*)d_in[23];
    const float* aw1c     = (const float*)d_in[24];
    const float* aw2c     = (const float*)d_in[25];
    const float* avc      = (const float*)d_in[26];

    float* out      = (float*)d_out;
    float* result   = out;                       // 16 x 20000
    float* ha_out   = out + (size_t)Bz * Vz;     // 16 x 512
    float* h1n      = ha_out + Bz * Hz;
    float* c1n      = h1n + Bz * Hz;
    float* h2n      = c1n + Bz * Hz;
    float* c2n      = h2n + Bz * Hz;

    // workspace layout (floats)
    float* ws       = (float*)d_ws;
    float* emb_att  = ws;                         // 12992
    float* q        = ws + 12992;                 // 16384
    float* context  = ws + 29376;                 // 8192
    float* align_c  = ws + 37568;                 // 9600
    float* sw       = ws + 47168;                 // 16
    float* rowmax   = ws + 47184;                 // 16
    float* rowsum   = ws + 47200;                 // 16
    float* gl       = ws + 47216;                 // 320000
    float* S        = ws + 367232;                // 9600*1024
    __bf16* enc_bf  = (__bf16*)(ws + 10197632);   // 9600*512 bf16
    __bf16* Wt      = (__bf16*)(ws + 12655232);   // 1024*512 bf16

    // data prep
    emb_kernel<<<(Bz * XDIM1 + 255) / 256, 256, 0, stream>>>(x, emb_tab, lha, emb_att);
    cvt_bf16_kernel<<<(BT * Hz + 255) / 256, 256, 0, stream>>>(enc_outs, enc_bf, BT * Hz);
    build_wt_kernel<<<(N2H * Hz + 255) / 256, 256, 0, stream>>>(aw2g, aw2c, Wt);

    // heavy WMMA GEMM (independent of the LSTM chain -> launched early)
    // 4800 waves = 1200 blocks x 4 waves
    attn_gemm_wmma<<<1200, 128, 0, stream>>>(enc_bf, Wt, S);

    // recurrent chain
    lstm_kernel<<<(Bz * Hz + 255) / 256, 256, 0, stream>>>(emb_att, XDIM1, h1, c1, k1, r1, b1, h1n, c1n);
    lstm_kernel<<<(Bz * Hz + 255) / 256, 256, 0, stream>>>(h1n, Hz, h2, c2, k2, r2, b2, h2n, c2n);
    qproj_kernel<<<(2 * Bz * Hz + 255) / 256, 256, 0, stream>>>(h2n, aw1g, aw1c, q);

    // attention (both heads), then decoder head
    attention_kernel<<<2 * Bz, 256, 0, stream>>>(S, q, avg, avc, enc_outs, context, align_c);
    hidden_att_kernel<<<(Bz * Hz + 255) / 256, 256, 0, stream>>>(context, h2n, Wh, bh, ha_out);
    switch_kernel<<<Bz, 256, 0, stream>>>(ha_out, Wf, bfp, sw);
    gen_logits_kernel<<<(Vz + 127) / 128, 128, 0, stream>>>(ha_out, Wg, bg, gl);
    softmax_stats_kernel<<<Bz, 256, 0, stream>>>(gl, rowmax, rowsum);

    dim3 fgrid((Vz + 255) / 256, Bz);
    final_kernel<<<fgrid, 256, 0, stream>>>(gl, rowmax, rowsum, sw, align_c, enc_ins, result);
}
